// GRUSkipCell_2044404432950
// MI455X (gfx1250) — matrix operand order repaired and checked by reading, not compiled, not run on hardware
//
#include <hip/hip_runtime.h>

// GRU cell, two-phase, for gfx1250 (MI455X).
// Phase 1: convert X,H -> bf16 row-major; Wk,Wr -> bf16 TRANSPOSED (N3 x D).
// Phase 2: fused dual-GEMM on V_WMMA_F32_16X16X32_BF16, fragments loaded
//          directly from global memory (layout-exact b128 loads), + GRU gates.

typedef __attribute__((ext_vector_type(16))) __bf16 v16bf;
typedef __attribute__((ext_vector_type(8)))  float  v8f;

#define Bsz 8192
#define Dsz 1024
#define Usz 1024
#define N3  3072

__device__ __forceinline__ unsigned short bf(float x) {
  return __builtin_bit_cast(unsigned short, (__bf16)x);
}
__device__ __forceinline__ unsigned int pk2(float x, float y) {
  return (unsigned int)bf(x) | ((unsigned int)bf(y) << 16);
}

// ---------------- Phase 1 kernels ----------------

__global__ __launch_bounds__(256)
void cvt_bf16_kernel(const float* __restrict__ in, unsigned short* __restrict__ out) {
  const size_t i = ((size_t)blockIdx.x * 256 + threadIdx.x) * 8;
  const float4 a = *reinterpret_cast<const float4*>(in + i);
  const float4 b = *reinterpret_cast<const float4*>(in + i + 4);
  uint4 p;
  p.x = pk2(a.x, a.y); p.y = pk2(a.z, a.w);
  p.z = pk2(b.x, b.y); p.w = pk2(b.z, b.w);
  *reinterpret_cast<uint4*>(out + i) = p;
}

// in: (Dsz, N3) f32  ->  out: (N3, Dsz) bf16  (32x32 LDS tile transpose)
__global__ __launch_bounds__(256)
void transpose_bf16_kernel(const float* __restrict__ in, unsigned short* __restrict__ out) {
  __shared__ unsigned short t[32][33];
  const int c0 = blockIdx.x * 32;               // column in 'in'  (0..N3)
  const int r0 = blockIdx.y * 32;               // row    in 'in'  (0..Dsz)
  const int tx = threadIdx.x, ty = threadIdx.y; // block (32, 8)
#pragma unroll
  for (int i = 0; i < 4; ++i)
    t[ty + i * 8][tx] = bf(in[(size_t)(r0 + ty + i * 8) * N3 + c0 + tx]);
  __syncthreads();
#pragma unroll
  for (int i = 0; i < 4; ++i)
    out[(size_t)(c0 + ty + i * 8) * Dsz + r0 + tx] = t[tx][ty + i * 8];
}

// ---------------- Phase 2: GEMM + gates ----------------

union FragU { v16bf v; uint4 q[2]; };

// A operand (16x32 bf16): lane holds row (lane&15); kb=(lane>>4)*16:
// v0-7 = K kb..kb+15 contiguous
__device__ __forceinline__ v16bf ldA(const unsigned short* p) {
  FragU u;
  u.q[0] = *reinterpret_cast<const uint4*>(p);
  u.q[1] = *reinterpret_cast<const uint4*>(p + 8);
  return u.v;
}
// B operand (32x16 bf16): lane holds col (lane&15); kb=(lane>>4)*16:
// v0-7 = K kb..kb+15 contiguous (from K-major transposed weights)
__device__ __forceinline__ v16bf ldB(const unsigned short* p) {
  FragU u;
  u.q[0] = *reinterpret_cast<const uint4*>(p);
  u.q[1] = *reinterpret_cast<const uint4*>(p + 8);
  return u.v;
}

#define WMMA(A, Bf, C) \
  __builtin_amdgcn_wmma_f32_16x16x32_bf16(false, (A), false, (Bf), (short)0, (C), false, false)

__device__ __forceinline__ float hsig(float x) {
  return fminf(fmaxf(__builtin_fmaf(0.2f, x, 0.5f), 0.0f), 1.0f);
}
__device__ __forceinline__ float fast_tanh(float x) {
  const float e = __expf(-2.0f * fabsf(x));          // v_exp_f32 (trans)
  const float t = __fdividef(1.0f - e, 1.0f + e);    // v_rcp_f32 + mul
  return copysignf(t, x);
}

__global__ __launch_bounds__(256)
void gru_gemm_wmma(const unsigned short* __restrict__ Xb,   // (8192,1024) bf16
                   const unsigned short* __restrict__ Hb,   // (8192,1024) bf16
                   const unsigned short* __restrict__ WkT,  // (3072,1024) bf16 (transposed)
                   const unsigned short* __restrict__ WrT,  // (3072,1024) bf16 (transposed)
                   const float* __restrict__ H,             // f32 h_tm1
                   const float* __restrict__ bias,          // (2,3072) f32
                   float* __restrict__ out) {               // (8192,1024) f32
  const int tid = threadIdx.x;
  const int lane = tid & 31;
  const int wave = tid >> 5;
  const int wrow = wave >> 1;                    // 0..3
  const int wcol = wave & 1;                     // 0..1
  const int rowBase = blockIdx.y * 128 + wrow * 32;  // wave: 32 rows
  const int colBase = blockIdx.x * 32 + wcol * 16;   // wave: 16 cols

  const int ml  = lane & 15;
  const int akb = (lane >> 4) * 16;
  const int bkb = (lane >> 4) * 16;

  // A-fragment base pointers (two 16-row sub-tiles each for X and H)
  const unsigned short* pX0 = Xb + (size_t)(rowBase + ml) * Dsz + akb;
  const unsigned short* pX1 = pX0 + 16 * Dsz;
  const unsigned short* pH0 = Hb + (size_t)(rowBase + ml) * Dsz + akb;
  const unsigned short* pH1 = pH0 + 16 * Dsz;
  // B-fragment base pointers: row (g*Usz + n) of transposed weights
  const unsigned short* pKz = WkT + (size_t)(0 * Usz + colBase + ml) * Dsz + bkb;
  const unsigned short* pKr = WkT + (size_t)(1 * Usz + colBase + ml) * Dsz + bkb;
  const unsigned short* pKh = WkT + (size_t)(2 * Usz + colBase + ml) * Dsz + bkb;
  const unsigned short* pRz = WrT + (size_t)(0 * Usz + colBase + ml) * Dsz + bkb;
  const unsigned short* pRr = WrT + (size_t)(1 * Usz + colBase + ml) * Dsz + bkb;
  const unsigned short* pRh = WrT + (size_t)(2 * Usz + colBase + ml) * Dsz + bkb;

  v8f aXZ[2] = {}, aXR[2] = {}, aXH[2] = {};
  v8f aRZ[2] = {}, aRR[2] = {}, aRH[2] = {};

  for (int k0 = 0; k0 < Dsz; k0 += 32) {
    const v16bf x0 = ldA(pX0 + k0);
    const v16bf x1 = ldA(pX1 + k0);
    const v16bf h0 = ldA(pH0 + k0);
    const v16bf h1 = ldA(pH1 + k0);

    v16bf bz = ldB(pKz + k0);
    v16bf br = ldB(pKr + k0);
    v16bf bh = ldB(pKh + k0);
    aXZ[0] = WMMA(x0, bz, aXZ[0]);  aXZ[1] = WMMA(x1, bz, aXZ[1]);
    aXR[0] = WMMA(x0, br, aXR[0]);  aXR[1] = WMMA(x1, br, aXR[1]);
    aXH[0] = WMMA(x0, bh, aXH[0]);  aXH[1] = WMMA(x1, bh, aXH[1]);

    bz = ldB(pRz + k0);
    br = ldB(pRr + k0);
    bh = ldB(pRh + k0);
    aRZ[0] = WMMA(h0, bz, aRZ[0]);  aRZ[1] = WMMA(h1, bz, aRZ[1]);
    aRR[0] = WMMA(h0, br, aRR[0]);  aRR[1] = WMMA(h1, br, aRR[1]);
    aRH[0] = WMMA(h0, bh, aRH[0]);  aRH[1] = WMMA(h1, bh, aRH[1]);
  }

  // ---- epilogue: bias + gates + blend ----
  const int n = colBase + ml;
  const float b0z = bias[n];
  const float b0r = bias[n + Usz];
  const float b0h = bias[n + 2 * Usz];
  const float b1z = bias[N3 + n];
  const float b1r = bias[N3 + n + Usz];
  const float b1h = bias[N3 + n + 2 * Usz];
  const int mhalf = (lane >> 4) * 8;

#pragma unroll
  for (int t = 0; t < 2; ++t) {
#pragma unroll
    for (int r = 0; r < 8; ++r) {
      const int m = rowBase + t * 16 + r + mhalf;   // C/D: v[r], lanes16-31 -> M=r+8
      const float z  = hsig((aXZ[t][r] + b0z) + (aRZ[t][r] + b1z));
      const float rg = hsig((aXR[t][r] + b0r) + (aRR[t][r] + b1r));
      const float hh = fast_tanh(__builtin_fmaf(rg, aRH[t][r] + b1h, aXH[t][r] + b0h));
      const float hp = H[(size_t)m * Usz + n];
      out[(size_t)m * Usz + n] = __builtin_fmaf(z, hp - hh, hh);  // z*hp+(1-z)*hh
    }
  }
}

// ---------------- launch ----------------

extern "C" void kernel_launch(void* const* d_in, const int* in_sizes, int n_in,
                              void* d_out, int out_size, void* d_ws, size_t ws_size,
                              hipStream_t stream) {
  const float* X  = (const float*)d_in[0];  // (8192,1024)
  const float* H  = (const float*)d_in[1];  // (8192,1024)
  const float* Wk = (const float*)d_in[2];  // (1024,3072)
  const float* Wr = (const float*)d_in[3];  // (1024,3072)
  const float* b  = (const float*)d_in[4];  // (2,3072)
  float* out = (float*)d_out;

  unsigned short* Xb  = (unsigned short*)d_ws;                 // 16 MB
  unsigned short* Hb  = Xb  + (size_t)Bsz * Dsz;               // 16 MB
  unsigned short* WkT = Hb  + (size_t)Bsz * Usz;               //  6 MB
  unsigned short* WrT = WkT + (size_t)N3 * Dsz;                //  6 MB (total 44 MB)

  cvt_bf16_kernel<<<(Bsz * Dsz) / (256 * 8), 256, 0, stream>>>(X, Xb);
  cvt_bf16_kernel<<<(Bsz * Usz) / (256 * 8), 256, 0, stream>>>(H, Hb);
  transpose_bf16_kernel<<<dim3(N3 / 32, Dsz / 32), dim3(32, 8), 0, stream>>>(Wk, WkT);
  transpose_bf16_kernel<<<dim3(N3 / 32, Dsz / 32), dim3(32, 8), 0, stream>>>(Wr, WrT);

  gru_gemm_wmma<<<dim3(Usz / 32, Bsz / 128), 256, 0, stream>>>(Xb, Hb, WkT, WrT, H, b, out);
}
